// Mamba_20392504721951
// MI455X (gfx1250) — matrix operand order repaired and checked by reading, not compiled, not run on hardware
//
#include <hip/hip_runtime.h>

// ---------------- problem dimensions ----------------
#define NB   64      // batch
#define NV   30      // sequence (joint) length
#define ND   640     // d_model
#define DIN  1280    // d_inner
#define NDS  16      // d_state
#define NDTR 40      // dt_rank
#define NDC  4       // d_conv
#define NNL  4       // n_layer
#define NM   (NB*NV) // 1920 rows
#define DBLP 80      // padded x_proj output width (72 -> 80)
#define DTKP 64      // padded dt_proj K (40 -> 64)

typedef __attribute__((ext_vector_type(8)))  unsigned short us8;
typedef __attribute__((ext_vector_type(16))) unsigned short us16;
typedef __attribute__((ext_vector_type(16))) __bf16        v16bf;
typedef __attribute__((ext_vector_type(8)))  float         v8f;

__device__ __forceinline__ unsigned short f2bf(float f) {
    unsigned int u = __builtin_bit_cast(unsigned int, f);
    unsigned int r = u + 0x7fffu + ((u >> 16) & 1u);   // round-to-nearest-even
    return (unsigned short)(r >> 16);
}

// ---------------- weight transpose + f32 -> bf16 convert ----------------
// W: [N, K] f32 row-major.  Wt: [Kpad, Npad] bf16, Wt[k*Npad+n] = W[n*K+k], zero padded.
__global__ void convert_wt_kernel(const float* __restrict__ W,
                                  unsigned short* __restrict__ Wt,
                                  int N, int K, int Npad, int Kpad) {
    int t = blockIdx.x * blockDim.x + threadIdx.x;
    if (t >= Kpad * Npad) return;
    int n = t % Npad, k = t / Npad;
    float v = (n < N && k < K) ? W[(size_t)n * K + k] : 0.f;
    Wt[t] = f2bf(v);
}

// ---------------- forward circular permute ----------------
__global__ void permute_kernel(const float* __restrict__ x,
                               const int* __restrict__ vs,
                               float* __restrict__ h) {
    int t = blockIdx.x * blockDim.x + threadIdx.x;
    if (t >= NM * ND) return;
    int d = t % ND, m = t / ND, v = m % NV, b = m / NV;
    int sv = (v + NV - vs[b]) % NV;
    h[t] = x[((size_t)(b * NV + sv)) * ND + d];
}

// ---------------- RMSNorm -> bf16 row kernel (one block per row) ----------------
__global__ __launch_bounds__(256)
void rmsnorm_bf16_kernel(const float* __restrict__ h,
                         const float* __restrict__ w,
                         unsigned short* __restrict__ out) {
    __shared__ float red[256];
    int m = blockIdx.x;
    const float* row = h + (size_t)m * ND;
    float s = 0.f;
    for (int d = threadIdx.x; d < ND; d += 256) { float x = row[d]; s += x * x; }
    red[threadIdx.x] = s; __syncthreads();
    for (int off = 128; off > 0; off >>= 1) {
        if (threadIdx.x < off) red[threadIdx.x] += red[threadIdx.x + off];
        __syncthreads();
    }
    float r = rsqrtf(red[0] / (float)ND + 1e-5f);
    for (int d = threadIdx.x; d < ND; d += 256)
        out[(size_t)m * ND + d] = f2bf(row[d] * r * w[d]);
}

// ---------------- fragment helpers (16x16x32 bf16 WMMA layouts) ----------------
// A-frag (16x32 bf16): lane L -> row L%16; elems 0-7: K = k0 + (L/16)*8 + i,
//                                     elems 8-15: K = k0 + 16 + (L/16)*8 + i.
// B-frag (32x16 bf16): lane L -> col n0+L%16; elems use the same K order as the A-frag.
// C/D f32: VGPR j -> row m0 + j + 8*(L/16), col n0 + (L%16).
__device__ __forceinline__ v16bf load_a_frag(const unsigned short* arow, int k0, int half) {
    us8 alo = *(const us8*)(arow + k0 + half * 8);
    us8 ahi = *(const us8*)(arow + k0 + 16 + half * 8);
    us16 au;
#pragma unroll
    for (int i = 0; i < 8; ++i) { au[i] = alo[i]; au[i + 8] = ahi[i]; }
    return __builtin_bit_cast(v16bf, au);
}

// ---------------- small WMMA GEMM (one 16x16 tile / wave) for N=80 ----------------
__global__ __launch_bounds__(128)
void gemm_bf16_wmma(const unsigned short* __restrict__ A,
                    const unsigned short* __restrict__ Wt,
                    const float* __restrict__ addsrc,
                    float* __restrict__ C,
                    int N, int K) {
    int wave = threadIdx.x >> 5;
    int lane = threadIdx.x & 31;
    int m0 = blockIdx.y * 64 + wave * 16;
    int n0 = blockIdx.x * 16;
    int half = lane >> 4;
    const unsigned short* arow  = A  + (size_t)(m0 + (lane & 15)) * K;
    const unsigned short* bbase = Wt + n0 + (lane & 15);

    v8f acc = {0.f, 0.f, 0.f, 0.f, 0.f, 0.f, 0.f, 0.f};
    for (int k0 = 0; k0 < K; k0 += 32) {
        v16bf a = load_a_frag(arow, k0, half);
        us16 bu;
#pragma unroll
        for (int i = 0; i < 8; ++i) {
            bu[i]     = bbase[(size_t)(k0 + half * 8 + i) * N];
            bu[i + 8] = bbase[(size_t)(k0 + 16 + half * 8 + i) * N];
        }
        v16bf b = __builtin_bit_cast(v16bf, bu);
        acc = __builtin_amdgcn_wmma_f32_16x16x32_bf16(
            false, a, false, b, (short)0, acc, false, false);
    }
    int col = n0 + (lane & 15);
#pragma unroll
    for (int j = 0; j < 8; ++j) {
        int row = m0 + j + 8 * half;
        size_t idx = (size_t)row * N + col;
        float v = acc[j];
        if (addsrc) v += addsrc[idx];
        C[idx] = v;
    }
}

// ---------------- blocked WMMA GEMM: wave computes 32x64 (2x4 tiles) ----------------
// Block = 128 threads = 4 waves in a 2x2 grid -> block tile 64 (M) x 128 (N).
// Per K-step: 2 A-frags + 4 B-frags feed 8 WMMAs -> ~21 FLOP/byte of fragment traffic.
// Requires N % 128 == 0, M % 64 == 0, K % 32 == 0.
__global__ __launch_bounds__(128)
void gemm_bf16_wmma_blk(const unsigned short* __restrict__ A,
                        const unsigned short* __restrict__ Wt,
                        const float* __restrict__ addsrc,
                        float* __restrict__ C,
                        int N, int K) {
    int wave = threadIdx.x >> 5;
    int lane = threadIdx.x & 31;
    int wm = wave >> 1, wn = wave & 1;
    int m0 = blockIdx.y * 64 + wm * 32;          // wave covers rows m0..m0+31
    int n0 = blockIdx.x * 128 + wn * 64;         // wave covers cols n0..n0+63
    int half = lane >> 4;

    const unsigned short* arow0 = A + (size_t)(m0 + (lane & 15)) * K;
    const unsigned short* arow1 = arow0 + (size_t)16 * K;
    const unsigned short* bbase = Wt + n0 + (lane & 15);

    v8f acc[2][4];
#pragma unroll
    for (int mt = 0; mt < 2; ++mt)
#pragma unroll
        for (int nt = 0; nt < 4; ++nt)
            acc[mt][nt] = (v8f){0.f, 0.f, 0.f, 0.f, 0.f, 0.f, 0.f, 0.f};

    for (int k0 = 0; k0 < K; k0 += 32) {
        v16bf a0 = load_a_frag(arow0, k0, half);
        v16bf a1 = load_a_frag(arow1, k0, half);
        v16bf b[4];
#pragma unroll
        for (int nt = 0; nt < 4; ++nt) {
            us16 bu;
#pragma unroll
            for (int i = 0; i < 8; ++i) {
                bu[i]     = bbase[(size_t)(k0 + half * 8 + i) * N + nt * 16];
                bu[i + 8] = bbase[(size_t)(k0 + 16 + half * 8 + i) * N + nt * 16];
            }
            b[nt] = __builtin_bit_cast(v16bf, bu);
        }
#pragma unroll
        for (int nt = 0; nt < 4; ++nt) {
            acc[0][nt] = __builtin_amdgcn_wmma_f32_16x16x32_bf16(
                false, a0, false, b[nt], (short)0, acc[0][nt], false, false);
            acc[1][nt] = __builtin_amdgcn_wmma_f32_16x16x32_bf16(
                false, a1, false, b[nt], (short)0, acc[1][nt], false, false);
        }
    }

    int colb = lane & 15;
#pragma unroll
    for (int mt = 0; mt < 2; ++mt) {
#pragma unroll
        for (int nt = 0; nt < 4; ++nt) {
#pragma unroll
            for (int j = 0; j < 8; ++j) {
                int row = m0 + mt * 16 + j + 8 * half;
                int col = n0 + nt * 16 + colb;
                size_t idx = (size_t)row * N + col;
                float v = acc[mt][nt][j];
                if (addsrc) v += addsrc[idx];
                C[idx] = v;
            }
        }
    }
}

// ---------------- causal depthwise conv (DC=4) + bias + SiLU ----------------
__global__ void conv_silu_kernel(const float* __restrict__ ur,   // [M, 2*DIN], u = cols 0..DIN-1
                                 const float* __restrict__ cw,   // [DIN, DC]
                                 const float* __restrict__ cb,   // [DIN]
                                 float* __restrict__ u,          // [M, DIN]
                                 unsigned short* __restrict__ ub) {
    int t = blockIdx.x * blockDim.x + threadIdx.x;
    if (t >= NM * DIN) return;
    int d = t % DIN, m = t / DIN, v = m % NV, b = m / NV;
    float s = cb[d];
#pragma unroll
    for (int j = 0; j < NDC; ++j) {
        int vv = v - (NDC - 1) + j;
        if (vv >= 0) s += cw[d * NDC + j] * ur[((size_t)(b * NV + vv)) * (2 * DIN) + d];
    }
    float val = s / (1.f + __expf(-s));   // silu
    u[t] = val;
    ub[t] = f2bf(val);
}

// ---------------- pack delta part of dbl into padded bf16 [M, 64] ----------------
__global__ void prep_delta_kernel(const float* __restrict__ dbl,
                                  unsigned short* __restrict__ dp) {
    int t = blockIdx.x * blockDim.x + threadIdx.x;
    if (t >= NM * DTKP) return;
    int k = t % DTKP, m = t / DTKP;
    float v = (k < NDTR) ? dbl[(size_t)m * DBLP + k] : 0.f;
    dp[t] = f2bf(v);
}

// ---------------- selective-scan: softplus, recurrence, C-dot, skip, gate ----------------
__global__ __launch_bounds__(256)
void scan_kernel(const float* __restrict__ ur,     // [M, 2*DIN] (res = cols DIN..)
                 const float* __restrict__ u,      // [M, DIN]
                 const float* __restrict__ dbl,    // [M, 80]  (B = 40..55, C = 56..71)
                 const float* __restrict__ dpre,   // [M, DIN] dt_proj output
                 const float* __restrict__ dt_b,   // [DIN]
                 const float* __restrict__ A_log,  // [DIN, DS]
                 const float* __restrict__ D_skip, // [DIN]
                 unsigned short* __restrict__ g) { // [M, DIN] bf16: (y+u*D)*silu(res)
    int t = blockIdx.x * blockDim.x + threadIdx.x;
    if (t >= NB * DIN) return;
    int d = t % DIN, b = t / DIN;
    float a[NDS], hs[NDS];
#pragma unroll
    for (int s = 0; s < NDS; ++s) { a[s] = -__expf(A_log[(size_t)d * NDS + s]); hs[s] = 0.f; }
    float bia = dt_b[d], dsk = D_skip[d];
    for (int v = 0; v < NV; ++v) {
        int m = b * NV + v;
        float xx = dpre[(size_t)m * DIN + d] + bia;
        float delta = (xx > 20.f) ? xx : log1pf(__expf(xx));   // softplus
        float uu = u[(size_t)m * DIN + d];
        float du = delta * uu;
        const float* bm = dbl + (size_t)m * DBLP + NDTR;
        const float* cm = bm + NDS;
        float y = 0.f;
#pragma unroll
        for (int s = 0; s < NDS; ++s) {
            hs[s] = __expf(delta * a[s]) * hs[s] + du * bm[s];
            y += hs[s] * cm[s];
        }
        float res  = ur[(size_t)m * (2 * DIN) + DIN + d];
        float gate = res / (1.f + __expf(-res));
        g[(size_t)m * DIN + d] = f2bf((y + uu * dsk) * gate);
    }
}

// ---------------- final RMSNorm + reverse circular permute ----------------
__global__ __launch_bounds__(256)
void final_kernel(const float* __restrict__ h,
                  const float* __restrict__ w,
                  const int* __restrict__ vs,
                  float* __restrict__ out) {
    __shared__ float red[256];
    int o = blockIdx.x;
    int b = o / NV, v = o % NV;
    int sv = (v + vs[b]) % NV;
    const float* row = h + (size_t)(b * NV + sv) * ND;
    float s = 0.f;
    for (int d = threadIdx.x; d < ND; d += 256) { float x = row[d]; s += x * x; }
    red[threadIdx.x] = s; __syncthreads();
    for (int off = 128; off > 0; off >>= 1) {
        if (threadIdx.x < off) red[threadIdx.x] += red[threadIdx.x + off];
        __syncthreads();
    }
    float r = rsqrtf(red[0] / (float)ND + 1e-5f);
    for (int d = threadIdx.x; d < ND; d += 256)
        out[(size_t)o * ND + d] = row[d] * r * w[d];
}

// ---------------- workspace layout ----------------
static constexpr size_t au256(size_t x) { return (x + 255) & ~(size_t)255; }
static constexpr size_t OFF_H     = 0;
static constexpr size_t OFF_XN    = au256(OFF_H     + (size_t)NM * ND * 4);
static constexpr size_t OFF_UR    = au256(OFF_XN    + (size_t)NM * ND * 2);
static constexpr size_t OFF_U     = au256(OFF_UR    + (size_t)NM * 2 * DIN * 4);
static constexpr size_t OFF_UB    = au256(OFF_U     + (size_t)NM * DIN * 4);
static constexpr size_t OFF_DBL   = au256(OFF_UB    + (size_t)NM * DIN * 2);
static constexpr size_t OFF_DP    = au256(OFF_DBL   + (size_t)NM * DBLP * 4);
static constexpr size_t OFF_DPRE  = au256(OFF_DP    + (size_t)NM * DTKP * 2);
static constexpr size_t OFF_G     = au256(OFF_DPRE  + (size_t)NM * DIN * 4);
static constexpr size_t OFF_INWT  = au256(OFF_G     + (size_t)NM * DIN * 2);
static constexpr size_t OFF_XPWT  = au256(OFF_INWT  + (size_t)NNL * ND * 2 * DIN * 2);
static constexpr size_t OFF_DTWT  = au256(OFF_XPWT  + (size_t)NNL * DIN * DBLP * 2);
static constexpr size_t OFF_OUTWT = au256(OFF_DTWT  + (size_t)NNL * DTKP * DIN * 2);

extern "C" void kernel_launch(void* const* d_in, const int* in_sizes, int n_in,
                              void* d_out, int out_size, void* d_ws, size_t ws_size,
                              hipStream_t stream) {
    (void)in_sizes; (void)n_in; (void)out_size; (void)ws_size;
    const float* x      = (const float*)d_in[0];
    const int*   vs     = (const int*)  d_in[1];
    const float* norm_w = (const float*)d_in[2];
    const float* in_w   = (const float*)d_in[3];
    const float* conv_w = (const float*)d_in[4];
    const float* conv_b = (const float*)d_in[5];
    const float* xp_w   = (const float*)d_in[6];
    const float* dt_w   = (const float*)d_in[7];
    const float* dt_b   = (const float*)d_in[8];
    const float* A_log  = (const float*)d_in[9];
    const float* D_skip = (const float*)d_in[10];
    const float* out_w  = (const float*)d_in[11];
    const float* norm_f = (const float*)d_in[12];
    float* out = (float*)d_out;

    char* ws = (char*)d_ws;
    float*          h     = (float*)         (ws + OFF_H);
    unsigned short* xn    = (unsigned short*)(ws + OFF_XN);
    float*          ur    = (float*)         (ws + OFF_UR);
    float*          u     = (float*)         (ws + OFF_U);
    unsigned short* ub    = (unsigned short*)(ws + OFF_UB);
    float*          dbl   = (float*)         (ws + OFF_DBL);
    unsigned short* dp    = (unsigned short*)(ws + OFF_DP);
    float*          dpre  = (float*)         (ws + OFF_DPRE);
    unsigned short* g     = (unsigned short*)(ws + OFF_G);
    unsigned short* inwt  = (unsigned short*)(ws + OFF_INWT);
    unsigned short* xpwt  = (unsigned short*)(ws + OFF_XPWT);
    unsigned short* dtwt  = (unsigned short*)(ws + OFF_DTWT);
    unsigned short* outwt = (unsigned short*)(ws + OFF_OUTWT);

    // ---- convert + transpose all weights to bf16 (deterministic, every call) ----
    for (int i = 0; i < NNL; ++i) {
        {   int tot = ND * 2 * DIN;   // in_proj: [2*DIN, ND] -> [ND, 2*DIN]
            convert_wt_kernel<<<(tot + 255) / 256, 256, 0, stream>>>(
                in_w + (size_t)i * 2 * DIN * ND, inwt + (size_t)i * ND * 2 * DIN,
                2 * DIN, ND, 2 * DIN, ND); }
        {   int tot = DIN * DBLP;     // x_proj: [72, DIN] -> [DIN, 80]
            convert_wt_kernel<<<(tot + 255) / 256, 256, 0, stream>>>(
                xp_w + (size_t)i * (NDTR + 2 * NDS) * DIN, xpwt + (size_t)i * DIN * DBLP,
                NDTR + 2 * NDS, DIN, DBLP, DIN); }
        {   int tot = DTKP * DIN;     // dt_proj: [DIN, 40] -> [64, DIN]
            convert_wt_kernel<<<(tot + 255) / 256, 256, 0, stream>>>(
                dt_w + (size_t)i * DIN * NDTR, dtwt + (size_t)i * DTKP * DIN,
                DIN, NDTR, DIN, DTKP); }
        {   int tot = DIN * ND;       // out_proj: [ND, DIN] -> [DIN, ND]
            convert_wt_kernel<<<(tot + 255) / 256, 256, 0, stream>>>(
                out_w + (size_t)i * ND * DIN, outwt + (size_t)i * DIN * ND,
                ND, DIN, ND, DIN); }
    }

    // ---- forward circular permute into residual stream h ----
    permute_kernel<<<(NM * ND + 255) / 256, 256, 0, stream>>>(x, vs, h);

    for (int i = 0; i < NNL; ++i) {
        rmsnorm_bf16_kernel<<<NM, 256, 0, stream>>>(h, norm_w + (size_t)i * ND, xn);
        // in_proj: N=2560 (%128 ok) -> blocked kernel
        gemm_bf16_wmma_blk<<<dim3((2 * DIN) / 128, NM / 64), 128, 0, stream>>>(
            xn, inwt + (size_t)i * ND * 2 * DIN, nullptr, ur, 2 * DIN, ND);
        conv_silu_kernel<<<(NM * DIN + 255) / 256, 256, 0, stream>>>(
            ur, conv_w + (size_t)i * DIN * NDC, conv_b + (size_t)i * DIN, u, ub);
        // x_proj: N=80 -> small kernel
        gemm_bf16_wmma<<<dim3(DBLP / 16, NM / 64), 128, 0, stream>>>(
            ub, xpwt + (size_t)i * DIN * DBLP, nullptr, dbl, DBLP, DIN);
        prep_delta_kernel<<<(NM * DTKP + 255) / 256, 256, 0, stream>>>(dbl, dp);
        // dt_proj: N=1280 (%128 ok) -> blocked kernel
        gemm_bf16_wmma_blk<<<dim3(DIN / 128, NM / 64), 128, 0, stream>>>(
            dp, dtwt + (size_t)i * DTKP * DIN, nullptr, dpre, DIN, DTKP);
        scan_kernel<<<(NB * DIN + 255) / 256, 256, 0, stream>>>(
            ur, u, dbl, dpre, dt_b + (size_t)i * DIN, A_log + (size_t)i * DIN * NDS,
            D_skip + (size_t)i * DIN, g);
        // out_proj: N=640 (%128 ok) -> blocked kernel, fused residual add
        gemm_bf16_wmma_blk<<<dim3(ND / 128, NM / 64), 128, 0, stream>>>(
            g, outwt + (size_t)i * DIN * ND, h, h, ND, DIN);
    }

    final_kernel<<<NM, 256, 0, stream>>>(h, norm_f, vs, out);
}
